// QuantLinear_30167850287309
// MI455X (gfx1250) — matrix operand order repaired and checked by reading, not compiled, not run on hardware
//
#include <hip/hip_runtime.h>
#include <hip/hip_bf16.h>

// ---------------------------------------------------------------------------
// W4A32 group-quantized GEMM for MI455X (gfx1250), wave32 + WMMA f16.
//   out[2048,11008] = x[2048,4096] @ dequant(qweight)[4096,11008]
// Compute-bound (AI ~1.26 kFLOP/B) -> drive v_wmma_f32_16x16x32_f16.
// Double-buffered LDS, one barrier per K-step.
// Dequant via packed-f16 magic numbers: 0x6400+nib == (f16)(1024+nib),
// exact pk_sub of 1024+z, single-rounded pk_mul by f16 scale.
// ---------------------------------------------------------------------------

typedef _Float16 v16h __attribute__((ext_vector_type(16)));
typedef _Float16 v8h  __attribute__((ext_vector_type(8)));
typedef _Float16 v4h  __attribute__((ext_vector_type(4)));
typedef _Float16 v2h  __attribute__((ext_vector_type(2)));
typedef float    v8f  __attribute__((ext_vector_type(8)));
typedef float    v4f  __attribute__((ext_vector_type(4)));

namespace {
constexpr int M_DIM = 2048;    // TOKENS
constexpr int K_DIM = 4096;    // IN_F
constexpr int N_DIM = 11008;   // OUT_F
constexpr int BM = 128;
constexpr int BN = 128;
constexpr int BK = 32;                      // one f16-WMMA K slab
constexpr int NSTEPS = K_DIM / BK;          // 128
constexpr int GROUP = 128;                  // quant group along K
}

union U32H2 { uint32_t u; v2h h; };

// Dequantize 8 nibbles of one packed int32 into 8 f16 weights using the
// 0x6400 magic-number trick: (1024+nib) - (1024+z) is exact in f16, then one
// packed multiply by the f16 scale. ~2x fewer VALU ops than the f32 path.
__device__ __forceinline__ v8h dequant8(uint32_t w, v2h hz, v2h sc) {
    constexpr uint32_t magic = 0x64006400u;       // f16x2 {1024, 1024}
    U32H2 c0, c1, c2, c3;
    c0.u = (w & 0xFu)          | ((w << 12) & 0x000F0000u) | magic;  // nib 0,1
    c1.u = ((w >> 8)  & 0xFu)  | ((w << 4)  & 0x000F0000u) | magic;  // nib 2,3
    c2.u = ((w >> 16) & 0xFu)  | ((w >> 4)  & 0x000F0000u) | magic;  // nib 4,5
    c3.u = ((w >> 24) & 0xFu)  | ((w >> 12) & 0x000F0000u) | magic;  // nib 6,7
    union { v8h v; U32H2 p[4]; } r;
    U32H2 t;
    t.h = (c0.h - hz) * sc; r.p[0] = t;
    t.h = (c1.h - hz) * sc; r.p[1] = t;
    t.h = (c2.h - hz) * sc; r.p[2] = t;
    t.h = (c3.h - hz) * sc; r.p[3] = t;
    return r.v;
}

__global__ __launch_bounds__(256)
void quantlinear_wmma_kernel(const float* __restrict__ x,
                             const int*   __restrict__ qweight,   // [K/8, N]
                             const int*   __restrict__ qzeros,    // [NGRP, N/8]
                             const int*   __restrict__ qscales,   // [NGRP, N] (uint8 values)
                             const float* __restrict__ qs_zero,   // [N]
                             const float* __restrict__ qs_scale,  // [N]
                             float*       __restrict__ out)       // [M, N]
{
    // Double-buffered LDS tiles. Bs is N-major / K-contiguous so each B
    // fragment is two aligned ds_load_b128 per lane.
    __shared__ _Float16 As[2][BM * BK];   // 2 x 8 KB
    __shared__ _Float16 Bs[2][BN * BK];   // 2 x 8 KB

    const int t    = threadIdx.x;
    const int wid  = t >> 5;          // wave id 0..7
    const int lane = t & 31;
    const int lm   = lane & 15;       // lane % 16
    const int lh   = lane >> 4;       // lane half 0/1

    const int wave_m = wid >> 2;      // 0..1  (64 rows each)
    const int wave_n = wid & 3;       // 0..3  (32 cols each)

    const int n0 = blockIdx.x * BN;
    const int m0 = blockIdx.y * BM;

    // --- dequant role: thread owns one n-column, two qweight rows per K-step
    const int n_off  = t & 127;                 // 0..127
    const int r_pair = t >> 7;                  // 0..1 -> rows {0,1} or {2,3}
    const int n_g    = n0 + n_off;              // global n for dequant
    const float zp = qs_zero[n_g];              // loop-invariant scale params
    const float ss = qs_scale[n_g];

    // --- A staging role: thread loads 4x float4, fixed (m,k) pattern per pass
    const int a_m = t >> 3;                     // row within pass block of 32
    const int a_k = (t & 7) << 2;               // 0,4,...,28

    v8f acc[4][2];
    #pragma unroll
    for (int tm = 0; tm < 4; ++tm)
        #pragma unroll
        for (int tn = 0; tn < 2; ++tn)
            acc[tm][tn] = (v8f)0.0f;

    // Rolling per-group dequant params (packed f16) for this thread's column.
    v2h sc2 = (v2h)0, hz2 = (v2h)0;
    auto group_params = [&](int g) {
        const float sc = ((float)qscales[g * N_DIM + n_g] - zp) * ss;
        const _Float16 sch = (_Float16)sc;
        sc2[0] = sch; sc2[1] = sch;
        const int zw = qzeros[g * (N_DIM / 8) + (n_g >> 3)];
        const int z  = ((zw >> ((n_g & 7) << 2)) & 0xF) + 1;      // 1..16
        U32H2 hzc; hzc.u = (uint32_t)(0x6400 + z) * 0x00010001u;  // f16x2 {1024+z,1024+z}
        hz2 = hzc.h;
        // warm L2 for next group's packed weights (global_prefetch_b8)
        if (g + 1 < K_DIM / GROUP)
            __builtin_prefetch(qweight + (size_t)((g + 1) * (GROUP / 8)) * N_DIM + n_g, 0, 1);
    };

    // ---- prologue: stage step 0 into buffer 0 ----
    group_params(0);
    {
        #pragma unroll
        for (int p = 0; p < 4; ++p) {
            const int m = p * 32 + a_m;
            const v4f xv = *(const v4f*)(x + (size_t)(m0 + m) * K_DIM + a_k);
            v4h hv;
            hv[0] = (_Float16)xv[0]; hv[1] = (_Float16)xv[1];
            hv[2] = (_Float16)xv[2]; hv[3] = (_Float16)xv[3];
            *(v4h*)(&As[0][m * BK + a_k]) = hv;
        }
        #pragma unroll
        for (int i = 0; i < 2; ++i) {
            const int rr = r_pair * 2 + i;
            const uint32_t w = (uint32_t)qweight[(size_t)rr * N_DIM + n_g];
            *(v8h*)(&Bs[0][n_off * BK + rr * 8]) = dequant8(w, hz2, sc2);
        }
    }
    __syncthreads();

    // ---- main loop: one barrier per K-step, LDS double-buffered ----
    for (int step = 0; step < NSTEPS; ++step) {
        const int buf   = step & 1;
        const int snext = step + 1;
        const bool have_next = (snext < NSTEPS);

        // (1) issue next step's global loads early (latency hidden under WMMA)
        v4f xv[4];
        uint32_t wv[2];
        if (have_next) {
            if ((snext & 3) == 0) group_params(snext >> 2);  // new quant group
            const int k0n = snext * BK;
            #pragma unroll
            for (int p = 0; p < 4; ++p) {
                const int m = p * 32 + a_m;
                xv[p] = *(const v4f*)(x + (size_t)(m0 + m) * K_DIM + k0n + a_k);
            }
            const int rbase = k0n >> 3;
            #pragma unroll
            for (int i = 0; i < 2; ++i)
                wv[i] = (uint32_t)qweight[(size_t)(rbase + r_pair * 2 + i) * N_DIM + n_g];
        }

        // (2) fragment ds loads from current buffer (ISA 7.12.2 layouts)
        v16h afrag[4], bfrag[2];
        #pragma unroll
        for (int tm = 0; tm < 4; ++tm) {
            const int m = wave_m * 64 + tm * 16 + lm;
            union { v16h v; v8h h[2]; } u;
            // A 16x32 f16: halves 0..7 -> K = 8*lh..8*lh+7; 8..15 -> +16
            u.h[0] = *(const v8h*)(&As[buf][m * BK + 8 * lh]);
            u.h[1] = *(const v8h*)(&As[buf][m * BK + 16 + 8 * lh]);
            afrag[tm] = u.v;
        }
        #pragma unroll
        for (int tn = 0; tn < 2; ++tn) {
            const int n = wave_n * 32 + tn * 16 + lm;
            union { v16h v; v8h h[2]; } u;
            // B 32x16 f16: halves 0..7 -> K = 8*lh..8*lh+7; 8..15 -> +16
            u.h[0] = *(const v8h*)(&Bs[buf][n * BK + 8 * lh]);
            u.h[1] = *(const v8h*)(&Bs[buf][n * BK + 16 + 8 * lh]);
            bfrag[tn] = u.v;
        }

        // (3) convert / dequant into the other buffer
        if (have_next) {
            #pragma unroll
            for (int p = 0; p < 4; ++p) {
                const int m = p * 32 + a_m;
                v4h hv;
                hv[0] = (_Float16)xv[p][0]; hv[1] = (_Float16)xv[p][1];
                hv[2] = (_Float16)xv[p][2]; hv[3] = (_Float16)xv[p][3];
                *(v4h*)(&As[buf ^ 1][m * BK + a_k]) = hv;
            }
            #pragma unroll
            for (int i = 0; i < 2; ++i) {
                const int rr = r_pair * 2 + i;
                *(v8h*)(&Bs[buf ^ 1][n_off * BK + rr * 8]) = dequant8(wv[i], hz2, sc2);
            }
        }

        // (4) matrix ops
        #pragma unroll
        for (int tm = 0; tm < 4; ++tm)
            #pragma unroll
            for (int tn = 0; tn < 2; ++tn)
                acc[tm][tn] = __builtin_amdgcn_wmma_f32_16x16x32_f16(
                    /*neg_a=*/false, afrag[tm],
                    /*neg_b=*/false, bfrag[tn],
                    /*c_mod=*/(short)0, acc[tm][tn],
                    /*reuse_a=*/false, /*reuse_b=*/false);

        // (5) single barrier per step
        __syncthreads();
    }

    // ---- store C: VGPR v of a 16x16 f32 tile holds M = v + 8*lh, N = lm ----
    #pragma unroll
    for (int tm = 0; tm < 4; ++tm) {
        const int mb = m0 + wave_m * 64 + tm * 16 + 8 * lh;
        #pragma unroll
        for (int tn = 0; tn < 2; ++tn) {
            const int n = n0 + wave_n * 32 + tn * 16 + lm;
            #pragma unroll
            for (int v = 0; v < 8; ++v)
                out[(size_t)(mb + v) * N_DIM + n] = acc[tm][tn][v];
        }
    }
}

extern "C" void kernel_launch(void* const* d_in, const int* in_sizes, int n_in,
                              void* d_out, int out_size, void* d_ws, size_t ws_size,
                              hipStream_t stream) {
    (void)in_sizes; (void)n_in; (void)out_size; (void)d_ws; (void)ws_size;
    const float* x        = (const float*)d_in[0];
    const int*   qweight  = (const int*)d_in[1];
    const int*   qzeros   = (const int*)d_in[2];
    const int*   qscales  = (const int*)d_in[3];
    const float* qs_zero  = (const float*)d_in[4];
    const float* qs_scale = (const float*)d_in[5];
    // d_in[6] = g_idx is the identity grouping (k / 128); computed inline.
    float* out = (float*)d_out;

    dim3 grid(N_DIM / BN, M_DIM / BM);   // 86 x 16
    quantlinear_wmma_kernel<<<grid, dim3(256), 0, stream>>>(
        x, qweight, qzeros, qscales, qs_zero, qs_scale, out);
}
